// SequenceLayer_35545149341914
// MI455X (gfx1250) — compile-verified, matrix operand order checked by reading, not run on hardware
//
#include <hip/hip_runtime.h>
#include <hip/hip_bf16.h>
#include <math.h>

typedef __attribute__((ext_vector_type(16))) __bf16 v16bf;
typedef __attribute__((ext_vector_type(8)))  __bf16 v8bf;
typedef __attribute__((ext_vector_type(8)))  float  v8f;

#define L_DIM 2048
#define B_DIM 16
#define H_DIM 512
#define P_DIM 256
#define M_DIM (L_DIM * B_DIM)   // 32768 rows
#define W_DIM 512               // K and N for all three GEMMs
#define NT_DIM (W_DIM / 16)     // 32 n-tiles
#define NSEG 32
#define SEGLEN (L_DIM / NSEG)   // 64
#define EPSV 1e-6f

// Packed-B index: B matrices are constant (512x512, 512KB) so we store them in
// exact WMMA B-fragment order: [kblock][ntile][lane][16 bf16] -> each lane's
// fragment is one contiguous 32-byte chunk (2x global_load_b128, L2-resident).
__device__ __forceinline__ size_t bpack_idx(int k, int n) {
    int kk = k >> 5, kr = k & 31;
    int nn = n >> 4, nl = n & 15;
    int half, e;
    if (kr < 16) { half = kr >> 3; e = kr & 7; }
    else         { half = (kr - 16) >> 3; e = 8 + ((kr - 16) & 7); }
    int lane = half * 16 + nl;
    return (((size_t)kk * NT_DIM + nn) * 32 + lane) * 16 + e;
}

// ---------------- prep: Lambda_bar and coef = (Lam_bar-1)/Lam ----------------
__global__ void prep_scalars(const float* __restrict__ Lre, const float* __restrict__ Lim,
                             const float* __restrict__ logstep,
                             float* __restrict__ lam_re, float* __restrict__ lam_im,
                             float* __restrict__ coef_re, float* __restrict__ coef_im) {
    int p = threadIdx.x;
    if (p < P_DIM) {
        float st = expf(logstep[p]);
        float lr = Lre[p], li = Lim[p];
        float er = expf(lr * st);
        float br = er * cosf(li * st);
        float bi = er * sinf(li * st);
        lam_re[p] = br; lam_im[p] = bi;
        float nr = br - 1.0f, ni = bi;
        float den = lr * lr + li * li;
        coef_re[p] = (nr * lr + ni * li) / den;
        coef_im[p] = (ni * lr - nr * li) / den;
    }
}

// Bmat1 logical [k=h][n]: n<256 -> Re(B_bar[n][h]) ; n>=256 -> Im(B_bar[n-256][h])
__global__ void prep_b1(const float* __restrict__ coef_re, const float* __restrict__ coef_im,
                        const float* __restrict__ B_re, const float* __restrict__ B_im,
                        __bf16* __restrict__ Bp) {
    int idx = blockIdx.x * 256 + threadIdx.x;   // over P*H
    int p = idx >> 9, h = idx & 511;
    float cr = coef_re[p], ci = coef_im[p];
    float b0r = B_re[p * H_DIM + h], b0i = B_im[p * H_DIM + h];
    float br = cr * b0r - ci * b0i;
    float bi = cr * b0i + ci * b0r;
    Bp[bpack_idx(h, p)]         = (__bf16)br;
    Bp[bpack_idx(h, P_DIM + p)] = (__bf16)bi;
}

// Bmat2 logical [k][n=h]: k<256 -> 2*C_re[h][k] ; k>=256 -> -2*C_im[h][k-256]
__global__ void prep_b2(const float* __restrict__ C_re, const float* __restrict__ C_im,
                        __bf16* __restrict__ Bp) {
    int idx = blockIdx.x * 256 + threadIdx.x;   // over 512*512
    int k = idx >> 9, n = idx & 511;
    float v = (k < P_DIM) ? (2.0f * C_re[(size_t)n * P_DIM + k])
                          : (-2.0f * C_im[(size_t)n * P_DIM + (k - P_DIM)]);
    Bp[bpack_idx(k, n)] = (__bf16)v;
}

__global__ void prep_b3(const float* __restrict__ w, __bf16* __restrict__ Bp) {
    int idx = blockIdx.x * 256 + threadIdx.x;   // over 512*512
    int k = idx >> 9, n = idx & 511;
    Bp[bpack_idx(k, n)] = (__bf16)w[idx];
}

// ---------------- LayerNorm (one block per row of 512) ----------------
template <bool GELU>
__global__ __launch_bounds__(256) void ln_kernel(const float* __restrict__ in,
                                                 const float* __restrict__ sc,
                                                 const float* __restrict__ bs,
                                                 float* __restrict__ of32,
                                                 __bf16* __restrict__ ob16) {
    int row = blockIdx.x;
    int tid = threadIdx.x;
    const float* r = in + (size_t)row * H_DIM;
    float v0 = r[tid], v1 = r[tid + 256];
    float s = v0 + v1, q = v0 * v0 + v1 * v1;
#pragma unroll
    for (int o = 16; o > 0; o >>= 1) {
        s += __shfl_xor(s, o, 32);
        q += __shfl_xor(q, o, 32);
    }
    __shared__ float ss[8], sq[8];
    int wv = tid >> 5;
    if ((tid & 31) == 0) { ss[wv] = s; sq[wv] = q; }
    __syncthreads();
    float S = 0.f, Q = 0.f;
#pragma unroll
    for (int i = 0; i < 8; i++) { S += ss[i]; Q += sq[i]; }
    float mu  = S * (1.0f / H_DIM);
    float var = Q * (1.0f / H_DIM) - mu * mu;
    float inv = rsqrtf(var + EPSV);
#pragma unroll
    for (int j = 0; j < 2; j++) {
        int h = tid + j * 256;
        float v = j ? v1 : v0;
        float y = (v - mu) * inv * sc[h] + bs[h];
        of32[(size_t)row * H_DIM + h] = y;
        float g = y;
        if (GELU) {
            float y3 = y * y * y;
            g = 0.5f * y * (1.0f + tanhf(0.7978845608f * (y + 0.044715f * y3)));
        }
        ob16[(size_t)row * H_DIM + h] = (__bf16)g;
    }
}

// ---------------- bf16 WMMA GEMM: C(MxN) = A(MxK) * B_packed(KxN) ----------------
// M=32768, N=K=512 fixed. Tiles: BM=128 (8 waves x 16 rows), BN=64, BK=32.
// A tile staged global->LDS with CDNA5 async B128 copies (ASYNCcnt), no VGPR
// round-trip; A fragments read as 2x ds_load_b128. B fragments read directly
// from fragment-packed global memory (2x global_load_b128, L2-resident).
// MODE 0: plain fp32 store.  MODE 2: C += Dvec[n]*xn[row*512+n] (ys epilogue).
template <int MODE>
__global__ __launch_bounds__(256) void gemm_bf16(const __bf16* __restrict__ A,
                                                 const __bf16* __restrict__ Bp,
                                                 float* __restrict__ Cout,
                                                 const float* __restrict__ Dvec,
                                                 const float* __restrict__ xn) {
    constexpr int BM = 128, BK = 32;
    __shared__ __align__(16) __bf16 As[BM][BK + 8];   // 80B row stride (16B aligned)
    int tid  = threadIdx.x;
    int wave = tid >> 5, lane = tid & 31;
    int bm = blockIdx.y * BM;
    int nn0 = blockIdx.x * 4;                 // first of 4 n-tiles for this block
    int mrow = lane & 15, half = lane >> 4;
    int m0 = wave * 16;

    v8f acc[4] = {};
    const v16bf* Bfrag = (const v16bf*)Bp;

    // per-thread staging slots: 512 x 16B chunks, 2 per thread
    int slot0 = tid, slot1 = 256 + tid;
    int row0 = slot0 >> 2, c40 = slot0 & 3;
    int row1 = slot1 >> 2, c41 = slot1 & 3;
    unsigned lds0 = (unsigned)(unsigned long long)(void*)&As[row0][c40 * 8];
    unsigned lds1 = (unsigned)(unsigned long long)(void*)&As[row1][c41 * 8];

    for (int kk = 0; kk < W_DIM / BK; kk++) {
        int k0 = kk * BK;
        // async global->LDS staging of the 128x32 A tile (2x B128 per thread)
        {
            unsigned long long g0 =
                (unsigned long long)(A + (size_t)(bm + row0) * W_DIM + k0 + c40 * 8);
            unsigned long long g1 =
                (unsigned long long)(A + (size_t)(bm + row1) * W_DIM + k0 + c41 * 8);
            asm volatile("global_load_async_to_lds_b128 %0, %1, off"
                         :: "v"(lds0), "v"(g0) : "memory");
            asm volatile("global_load_async_to_lds_b128 %0, %1, off"
                         :: "v"(lds1), "v"(g1) : "memory");
            asm volatile("s_wait_asynccnt 0x0" ::: "memory");
        }
        __syncthreads();

        // A fragment: lane's 16 values are two contiguous 8-elem (16B) chunks
        v8bf alo = *(const v8bf*)&As[m0 + mrow][half * 8];
        v8bf ahi = *(const v8bf*)&As[m0 + mrow][16 + half * 8];
        v16bf afr = __builtin_shufflevector(alo, ahi,
            0, 1, 2, 3, 4, 5, 6, 7, 8, 9, 10, 11, 12, 13, 14, 15);

#pragma unroll
        for (int nt = 0; nt < 4; nt++) {
            // B fragment: one contiguous 32B chunk per lane from packed global
            v16bf bfr = Bfrag[((size_t)kk * NT_DIM + nn0 + nt) * 32 + lane];
            acc[nt] = __builtin_amdgcn_wmma_f32_16x16x32_bf16(
                false, afr, false, bfr, (short)0, acc[nt], false, false);
        }
        __syncthreads();
    }

    // epilogue: C layout — lane%16 = N col, VGPR r = M (lanes 16-31 => M+8).
    // One base pointer per n-tile; r*512 folds into the 24-bit IOFFSET.
    int crow = bm + m0 + 8 * half;
#pragma unroll
    for (int nt = 0; nt < 4; nt++) {
        int n = (nn0 + nt) * 16 + mrow;
        float* cp = Cout + (size_t)crow * W_DIM + n;
        if constexpr (MODE == 2) {
            float dv = Dvec[n];
            const float* xp = xn + (size_t)crow * W_DIM + n;
#pragma unroll
            for (int r = 0; r < 8; r++)
                cp[r * W_DIM] = acc[nt][r] + dv * xp[r * W_DIM];
        } else {
#pragma unroll
            for (int r = 0; r < 8; r++)
                cp[r * W_DIM] = acc[nt][r];
        }
    }
}

// ---------------- chunked associative scan: h[l] = a[l]*h[l-1] + Bu[l] ----------------
__global__ __launch_bounds__(256) void scan_pass1(const float* __restrict__ Bu,
                                                  const unsigned char* __restrict__ drop,
                                                  const float* __restrict__ lam_re,
                                                  const float* __restrict__ lam_im,
                                                  const float* __restrict__ h0re,
                                                  const float* __restrict__ h0im,
                                                  float4* __restrict__ seg) {
    int t = blockIdx.x * 256 + threadIdx.x;
    int p = t & (P_DIM - 1);
    int b = (t >> 8) & (B_DIM - 1);
    int s = t >> 12;
    float lr = lam_re[p], li = lam_im[p];
    float Ar = 1.f, Ai = 0.f, hr = 0.f, hi = 0.f;
    int l0 = s * SEGLEN;
    for (int i = 0; i < SEGLEN; i++) {
        int l = l0 + i;
        float keep = drop[l * B_DIM + b] ? 0.f : 1.f;
        float ar = lr * keep, ai = li * keep;
        size_t rr = (size_t)(l * B_DIM + b) * W_DIM;
        float br = Bu[rr + p], bi = Bu[rr + P_DIM + p];
        if (l == 0) {
            float x0r = h0re[b * P_DIM + p], x0i = h0im[b * P_DIM + p];
            br += ar * x0r - ai * x0i;
            bi += ar * x0i + ai * x0r;
        }
        float nhr = ar * hr - ai * hi + br;
        float nhi = ar * hi + ai * hr + bi;
        hr = nhr; hi = nhi;
        float nAr = ar * Ar - ai * Ai;
        float nAi = ar * Ai + ai * Ar;
        Ar = nAr; Ai = nAi;
    }
    seg[(size_t)s * (B_DIM * P_DIM) + b * P_DIM + p] = make_float4(Ar, Ai, hr, hi);
}

__global__ __launch_bounds__(256) void scan_pass2(const float* __restrict__ Bu,
                                                  const unsigned char* __restrict__ drop,
                                                  const float* __restrict__ lam_re,
                                                  const float* __restrict__ lam_im,
                                                  const float* __restrict__ h0re,
                                                  const float* __restrict__ h0im,
                                                  const float4* __restrict__ seg,
                                                  __bf16* __restrict__ xs16,
                                                  float* __restrict__ hid_out) {
    int t = blockIdx.x * 256 + threadIdx.x;
    int p = t & (P_DIM - 1);
    int b = (t >> 8) & (B_DIM - 1);
    int s = t >> 12;
    float lr = lam_re[p], li = lam_im[p];
    // carry = scan of previous segment summaries
    float hr = 0.f, hi = 0.f;
    for (int j = 0; j < s; j++) {
        float4 sv = seg[(size_t)j * (B_DIM * P_DIM) + b * P_DIM + p];
        float nr = sv.x * hr - sv.y * hi + sv.z;
        float ni = sv.x * hi + sv.y * hr + sv.w;
        hr = nr; hi = ni;
    }
    int l0 = s * SEGLEN;
    for (int i = 0; i < SEGLEN; i++) {
        int l = l0 + i;
        float keep = drop[l * B_DIM + b] ? 0.f : 1.f;
        float ar = lr * keep, ai = li * keep;
        size_t rr = (size_t)(l * B_DIM + b) * W_DIM;
        float br = Bu[rr + p], bi = Bu[rr + P_DIM + p];
        if (l == 0) {
            float x0r = h0re[b * P_DIM + p], x0i = h0im[b * P_DIM + p];
            br += ar * x0r - ai * x0i;
            bi += ar * x0i + ai * x0r;
        }
        float nhr = ar * hr - ai * hi + br;
        float nhi = ar * hi + ai * hr + bi;
        hr = nhr; hi = nhi;
        xs16[rr + p]         = (__bf16)hr;
        xs16[rr + P_DIM + p] = (__bf16)hi;
        if (l == L_DIM - 1) {
            hid_out[b * P_DIM + p]                 = hr;
            hid_out[B_DIM * P_DIM + b * P_DIM + p] = hi;
        }
    }
}

// ---------------- final: out = x + y * sigmoid(z + b2) ----------------
__global__ __launch_bounds__(256) void final_kernel(const float* __restrict__ x,
                                                    const float* __restrict__ y,
                                                    const float* __restrict__ z,
                                                    const float* __restrict__ b2,
                                                    float* __restrict__ out) {
    size_t idx = (size_t)blockIdx.x * 256 + threadIdx.x;
    int h = idx & (W_DIM - 1);
    float tv = z[idx] + b2[h];
    float sg = 1.0f / (1.0f + expf(-tv));
    out[idx] = x[idx] + y[idx] * sg;
}

extern "C" void kernel_launch(void* const* d_in, const int* in_sizes, int n_in,
                              void* d_out, int out_size, void* d_ws, size_t ws_size,
                              hipStream_t stream) {
    (void)in_sizes; (void)n_in; (void)out_size; (void)ws_size;
    const float*         x        = (const float*)d_in[0];
    const unsigned char* drop     = (const unsigned char*)d_in[1];   // bool (1B)
    const float*         h0re     = (const float*)d_in[2];
    const float*         h0im     = (const float*)d_in[3];
    const float*         Lre      = (const float*)d_in[4];
    const float*         Lim      = (const float*)d_in[5];
    const float*         B_re     = (const float*)d_in[6];
    const float*         B_im     = (const float*)d_in[7];
    const float*         C_re     = (const float*)d_in[8];
    const float*         C_im     = (const float*)d_in[9];
    const float*         Dv       = (const float*)d_in[10];
    const float*         logstep  = (const float*)d_in[11];
    const float*         ln_scale = (const float*)d_in[12];
    const float*         ln_bias  = (const float*)d_in[13];
    const float*         out2_w   = (const float*)d_in[14];
    const float*         out2_b   = (const float*)d_in[15];
    float*               out      = (float*)d_out;

    char* ws = (char*)d_ws;
    const size_t MB = 1ull << 20;
    float*  bufA = (float*)(ws);              // xn_f32, later z      (64 MB)
    __bf16* bufB = (__bf16*)(ws + 64 * MB);   // xn_bf16, later x1_16 (32 MB)
    float*  bufC = (float*)(ws + 96 * MB);    // Bu, then ys, then y  (64 MB)
    __bf16* bufD = (__bf16*)(ws + 160 * MB);  // xs_bf16              (32 MB)
    char* sm = ws + 192 * MB;
    __bf16* Bmat1 = (__bf16*)sm; sm += 512 * 1024;
    __bf16* Bmat2 = (__bf16*)sm; sm += 512 * 1024;
    __bf16* Bmat3 = (__bf16*)sm; sm += 512 * 1024;
    float* lam_re  = (float*)sm; sm += 1024;
    float* lam_im  = (float*)sm; sm += 1024;
    float* coef_re = (float*)sm; sm += 1024;
    float* coef_im = (float*)sm; sm += 1024;
    float4* seg = (float4*)sm;                // 32*4096*16 = 2 MB

    prep_scalars<<<1, 256, 0, stream>>>(Lre, Lim, logstep, lam_re, lam_im, coef_re, coef_im);
    prep_b1<<<(P_DIM * H_DIM) / 256, 256, 0, stream>>>(coef_re, coef_im, B_re, B_im, Bmat1);
    prep_b2<<<(W_DIM * W_DIM) / 256, 256, 0, stream>>>(C_re, C_im, Bmat2);
    prep_b3<<<(W_DIM * W_DIM) / 256, 256, 0, stream>>>(out2_w, Bmat3);

    // xn = LN(x): fp32 (for D*xn) + bf16 (GEMM1 A)
    ln_kernel<false><<<M_DIM, 256, 0, stream>>>(x, ln_scale, ln_bias, bufA, bufB);

    dim3 ggrid(W_DIM / 64, M_DIM / 128);
    // GEMM1: Bu = xn @ [Bbar_re | Bbar_im]
    gemm_bf16<0><<<ggrid, 256, 0, stream>>>(bufB, Bmat1, bufC, nullptr, nullptr);

    // chunked scan (also emits new_hidden re/im into d_out[0 .. 2*B*P))
    scan_pass1<<<(NSEG * B_DIM * P_DIM) / 256, 256, 0, stream>>>(bufC, drop, lam_re, lam_im, h0re, h0im, seg);
    scan_pass2<<<(NSEG * B_DIM * P_DIM) / 256, 256, 0, stream>>>(bufC, drop, lam_re, lam_im, h0re, h0im, seg, bufD, out);

    // GEMM2: ys = [xs_re|xs_im] @ [2C_re; -2C_im] + D*xn   (overwrites Bu buffer)
    gemm_bf16<2><<<ggrid, 256, 0, stream>>>(bufD, Bmat2, bufC, Dv, bufA);

    // y = LN(ys) in-place fp32; x1 = gelu(y) bf16
    ln_kernel<true><<<M_DIM, 256, 0, stream>>>(bufC, ln_scale, ln_bias, bufC, bufB);

    // GEMM3: z = x1 @ out2_w  (reuses xn fp32 buffer)
    gemm_bf16<0><<<ggrid, 256, 0, stream>>>(bufB, Bmat3, bufA, nullptr, nullptr);

    // out = x + y * sigmoid(z + b)
    final_kernel<<<(M_DIM * W_DIM) / 256, 256, 0, stream>>>(x, bufC, bufA, out2_b, out + 2 * B_DIM * P_DIM);
}
